// SolarGNN_82351702933640
// MI455X (gfx1250) — compile-verified, matrix operand order checked by reading, not run on hardware
//
#include <hip/hip_runtime.h>
#include <hip/hip_bf16.h>

typedef __attribute__((ext_vector_type(16))) __bf16 v16bf;
typedef __attribute__((ext_vector_type(8)))  __bf16 v8bf;
typedef __attribute__((ext_vector_type(8)))  float  v8f;

// ---------------------------------------------------------------------------
// Degree / normalization kernels
// ---------------------------------------------------------------------------
__global__ void deg_init_kernel(float* __restrict__ deg, int n) {
    int i = blockIdx.x * blockDim.x + threadIdx.x;
    if (i < n) deg[i] = 1.0f;               // self-loop contributes 1
}

__global__ void deg_count_kernel(const long long* __restrict__ ei,
                                 float* __restrict__ deg, long long E) {
    long long i = (long long)blockIdx.x * blockDim.x + threadIdx.x;
    if (i < E) {
        long long d = ei[E + i];            // dst row of edge_index
        unsafeAtomicAdd(&deg[d], 1.0f);
    }
}

__global__ void dinv_kernel(float* __restrict__ deg, int n) {
    int i = blockIdx.x * blockDim.x + threadIdx.x;
    if (i < n) deg[i] = rsqrtf(deg[i]);     // deg >= 1 always
}

// ---------------------------------------------------------------------------
// Dense GEMM via v_wmma_f32_16x16x32_bf16 with W staged in LDS (bf16,
// transposed) so each lane's B fragment is two contiguous 16B LDS reads.
//
// One wave computes a 16(M) x 64(N) output tile; K looped in steps of 32.
// X: [n, kdim] row-major f32 (converted to bf16 on the fly)
// W: [kdim, 64] row-major f32  ->  LDS Wt[col][k] bf16
// H: [n, 64] f32 output
//
// A-fragment (16-bit A 16x32): lane L (M = L&15, half = L>>4),
//   elements 0..7  -> K = kb + 8*half + e
//   elements 8..15 -> K = kb + 16 + 8*half + (e-8)
// B-fragment mirrors A with N (col) in place of M.
// C/D: VGPR j -> row M = j + 8*half, col N = L&15.
// ---------------------------------------------------------------------------
__global__ void gcn_gemm_wmma_kernel(const float* __restrict__ X,
                                     const float* __restrict__ W,
                                     float* __restrict__ H,
                                     int n, int kdim) {
    __shared__ __bf16 Wt[64 * 128];          // max kdim = 128 -> 16 KB

    // ---- cooperative staging: Wt[c * kdim + k] = bf16(W[k * 64 + c])
    for (int idx = threadIdx.x; idx < kdim * 64; idx += blockDim.x) {
        int k = idx >> 6;
        int c = idx & 63;
        Wt[c * kdim + k] = (__bf16)W[idx];
    }
    __syncthreads();

    const int wave = threadIdx.x >> 5;
    const int lane = threadIdx.x & 31;
    const int half = lane >> 4;
    const int r    = lane & 15;
    const int row0 = blockIdx.x * 64 + wave * 16;   // wave-uniform

    if (row0 < n) {                                 // uniform: EXEC all-1s inside
        v8f acc[4];
        #pragma unroll
        for (int t = 0; t < 4; ++t) acc[t] = (v8f){0.f,0.f,0.f,0.f,0.f,0.f,0.f,0.f};

        const float* xrow = X + (size_t)(row0 + r) * (size_t)kdim;

        for (int kb = 0; kb < kdim; kb += 32) {
            // ---- A fragment: row (row0 + r), K in [kb, kb+32). 4x b128 loads.
            const float* ap = xrow + kb + 8 * half;
            float af[16];
            *(float4*)(af + 0)  = *(const float4*)(ap + 0);
            *(float4*)(af + 4)  = *(const float4*)(ap + 4);
            *(float4*)(af + 8)  = *(const float4*)(ap + 16);
            *(float4*)(af + 12) = *(const float4*)(ap + 20);
            v16bf a;
            #pragma unroll
            for (int e = 0; e < 16; ++e) a[e] = (__bf16)af[e];

            // ---- B fragments from LDS: 4 column tiles, 2x 16B reads each
            #pragma unroll
            for (int t = 0; t < 4; ++t) {
                const __bf16* bcol = &Wt[(size_t)(t * 16 + r) * kdim + kb + 8 * half];
                v8bf lo = *(const v8bf*)(bcol);        // K = kb+8h .. +7
                v8bf hi = *(const v8bf*)(bcol + 16);   // K = kb+16+8h .. +7
                v16bf b;
                #pragma unroll
                for (int e = 0; e < 8; ++e) { b[e] = lo[e]; b[8 + e] = hi[e]; }

                acc[t] = __builtin_amdgcn_wmma_f32_16x16x32_bf16(
                    false, a, false, b, (short)0, acc[t], false, false);
            }
        }

        // ---- store D (lanes 0..15 cover 16 consecutive cols -> coalesced)
        float* hout = H + (size_t)row0 * 64;
        #pragma unroll
        for (int j = 0; j < 8; ++j) {
            const int m = j + 8 * half;
            hout[(size_t)m * 64 +  0 + r] = acc[0][j];
            hout[(size_t)m * 64 + 16 + r] = acc[1][j];
            hout[(size_t)m * 64 + 32 + r] = acc[2][j];
            hout[(size_t)m * 64 + 48 + r] = acc[3][j];
        }
    }
}

// ---------------------------------------------------------------------------
// agg[i][:] = h[i][:] * dinv[i]^2   (self-loop seed; also initializes agg)
// float4 per thread: i indexes n*16 float4s.
// ---------------------------------------------------------------------------
__global__ void selfloop_kernel(const float* __restrict__ h,
                                const float* __restrict__ dinv,
                                float* __restrict__ agg, int n) {
    long long i = (long long)blockIdx.x * blockDim.x + threadIdx.x;
    if (i < (long long)n * 16) {
        int node = (int)(i >> 4);
        float di = dinv[node];
        float s  = di * di;
        float4 v = ((const float4*)h)[i];
        float4 o = make_float4(v.x * s, v.y * s, v.z * s, v.w * s);
        ((float4*)agg)[i] = o;
    }
}

// ---------------------------------------------------------------------------
// Edge scatter: 16 threads per edge, float4 gather + 4 f32 atomics each.
// agg[dst][f] += h[src][f] * dinv[src]*dinv[dst]
// ---------------------------------------------------------------------------
__global__ void edge_agg_kernel(const long long* __restrict__ ei,
                                const float* __restrict__ h,
                                const float* __restrict__ dinv,
                                float* __restrict__ agg, long long E) {
    long long e = (long long)blockIdx.x * 16 + (threadIdx.x >> 4);
    int f = (threadIdx.x & 15) << 2;
    if (e >= E) return;
    long long s = ei[e];
    long long d = ei[E + e];
    float nm = dinv[s] * dinv[d];
    float4 hv = *(const float4*)(h + s * 64 + f);
    float* ap = agg + d * 64 + f;
    unsafeAtomicAdd(ap + 0, hv.x * nm);
    unsafeAtomicAdd(ap + 1, hv.y * nm);
    unsafeAtomicAdd(ap + 2, hv.z * nm);
    unsafeAtomicAdd(ap + 3, hv.w * nm);
}

// ---------------------------------------------------------------------------
// out[i][:] = relu(agg[i][:] + b[:])  -- float4 per thread over n*16 float4s
// ---------------------------------------------------------------------------
__global__ void bias_relu_kernel(const float* __restrict__ agg,
                                 const float* __restrict__ b,
                                 float* __restrict__ out, int n) {
    long long i = (long long)blockIdx.x * blockDim.x + threadIdx.x;
    if (i < (long long)n * 16) {
        float4 bv = ((const float4*)b)[i & 15];
        float4 v  = ((const float4*)agg)[i];
        float4 o  = make_float4(fmaxf(v.x + bv.x, 0.f), fmaxf(v.y + bv.y, 0.f),
                                fmaxf(v.z + bv.z, 0.f), fmaxf(v.w + bv.w, 0.f));
        ((float4*)out)[i] = o;
    }
}

// ---------------------------------------------------------------------------
// Final projection: out[i] = dot(h[i,:64], W3) + b3. One wave per node.
// ---------------------------------------------------------------------------
__global__ void final_proj_kernel(const float* __restrict__ h,
                                  const float* __restrict__ W3,
                                  const float* __restrict__ b3,
                                  float* __restrict__ out, int n) {
    int node = (int)((long long)blockIdx.x * blockDim.x + threadIdx.x) >> 5;
    int lane = threadIdx.x & 31;
    if (node >= n) return;
    const float* row = h + (size_t)node * 64;
    float v = row[lane] * W3[lane] + row[lane + 32] * W3[lane + 32];
    #pragma unroll
    for (int off = 16; off > 0; off >>= 1) v += __shfl_xor(v, off, 32);
    if (lane == 0) out[node] = v + b3[0];
}

// ---------------------------------------------------------------------------
extern "C" void kernel_launch(void* const* d_in, const int* in_sizes, int n_in,
                              void* d_out, int out_size, void* d_ws, size_t ws_size,
                              hipStream_t stream) {
    const float*     x   = (const float*)d_in[0];
    const long long* ei  = (const long long*)d_in[1];
    const float*     W1  = (const float*)d_in[2];
    const float*     b1  = (const float*)d_in[3];
    const float*     W2  = (const float*)d_in[4];
    const float*     b2  = (const float*)d_in[5];
    const float*     W3  = (const float*)d_in[6];
    const float*     b3  = (const float*)d_in[7];
    float*           out = (float*)d_out;

    const int       N = in_sizes[0] / 128;
    const long long E = (long long)in_sizes[1] / 2;

    // workspace layout: dinv[N] | hbuf[N*64] | aggbuf[N*64]
    float* dinv   = (float*)d_ws;
    float* hbuf   = dinv + N;
    float* aggbuf = hbuf + (size_t)N * 64;

    const int T = 256;
    const int gN    = (N + T - 1) / T;
    const int gE    = (int)((E + T - 1) / T);
    const int gNF4  = (int)(((long long)N * 16 + T - 1) / T);   // float4 streams
    const int gEdge = (int)((E + 15) / 16);       // 16 edges x 16 threads per block
    const int gGemm = (N + 63) / 64;              // 4 waves x 16 rows per block
    const int gWave = (N + 7) / 8;                // 8 waves per block of 256

    // ---- symmetric normalization coefficients
    deg_init_kernel <<<gN,    T, 0, stream>>>(dinv, N);
    deg_count_kernel<<<gE,    T, 0, stream>>>(ei, dinv, E);
    dinv_kernel     <<<gN,    T, 0, stream>>>(dinv, N);

    // ---- layer 1: h1 = x @ W1 ; aggregate ; relu(+b1)
    gcn_gemm_wmma_kernel<<<gGemm, 128, 0, stream>>>(x, W1, hbuf, N, 128);
    selfloop_kernel     <<<gNF4,  T, 0, stream>>>(hbuf, dinv, aggbuf, N);
    edge_agg_kernel     <<<gEdge, T, 0, stream>>>(ei, hbuf, dinv, aggbuf, E);
    bias_relu_kernel    <<<gNF4,  T, 0, stream>>>(aggbuf, b1, hbuf, N);   // h1 -> hbuf

    // ---- layer 2: h2 = h1 @ W2 ; aggregate ; relu(+b2)
    gcn_gemm_wmma_kernel<<<gGemm, 128, 0, stream>>>(hbuf, W2, aggbuf, N, 64); // h2pre -> aggbuf
    selfloop_kernel     <<<gNF4,  T, 0, stream>>>(aggbuf, dinv, hbuf, N);     // agg2 -> hbuf
    edge_agg_kernel     <<<gEdge, T, 0, stream>>>(ei, aggbuf, dinv, hbuf, E);
    bias_relu_kernel    <<<gNF4,  T, 0, stream>>>(hbuf, b2, aggbuf, N);       // h2 -> aggbuf

    // ---- head: out = h2 @ W3 + b3
    final_proj_kernel   <<<gWave, T, 0, stream>>>(aggbuf, W3, b3, out, N);
}
